// NonLocalBlock_52261162057864
// MI455X (gfx1250) — matrix operand order repaired and checked by reading, not compiled, not run on hardware
//
#include <hip/hip_runtime.h>

#define BATCH 4
#define CIN   256
#define CB    128
#define NSP   4096  // H*W = 64*64

typedef __attribute__((ext_vector_type(16))) __bf16 v16bf;
typedef __attribute__((ext_vector_type(8)))  float  v8f;

__device__ __forceinline__ v8f wmma_bf16(v16bf a, v16bf b, v8f c) {
  return __builtin_amdgcn_wmma_f32_16x16x32_bf16(false, a, false, b, (short)0, c,
                                                 false, false);
}

// A fragment 16x32 (rows, K) from row-major bf16 tile; rs = row stride (elems).
// Layout (ISA 7.12.2): lane<16 -> row=lane, K {kb..kb+7, kb+16..kb+23} kb=0;
// lane>=16 -> kb=8.
__device__ __forceinline__ v16bf load_a_bf16(const __bf16* tile, int rs,
                                             int row0, int k0, int lane) {
  const __bf16* p = tile + (size_t)(row0 + (lane & 15)) * rs + k0 + ((lane >> 4) * 8);
  v16bf a;
#pragma unroll
  for (int e = 0; e < 8; ++e) a[e] = p[e];
#pragma unroll
  for (int e = 0; e < 8; ++e) a[8 + e] = p[16 + e];
  return a;
}

// Same but source is fp32 row-major (weights) converted to bf16 on load.
__device__ __forceinline__ v16bf load_a_f32(const float* tile, int rs,
                                            int row0, int k0, int lane) {
  const float* p = tile + (size_t)(row0 + (lane & 15)) * rs + k0 + ((lane >> 4) * 8);
  v16bf a;
#pragma unroll
  for (int e = 0; e < 8; ++e) a[e] = (__bf16)p[e];
#pragma unroll
  for (int e = 0; e < 8; ++e) a[8 + e] = (__bf16)p[16 + e];
  return a;
}

// B fragment 32x16 (K, cols) from a tile staged column-major: tileT[col][k],
// row stride rs. lane<16: col=lane, k {k0..k0+7, k0+16..k0+23}; lane>=16: kb=k0+8.
__device__ __forceinline__ v16bf load_b_t(const __bf16* tileT, int rs,
                                          int col0, int k0, int lane) {
  const __bf16* p = tileT + (size_t)(col0 + (lane & 15)) * rs + k0 + ((lane >> 4) * 8);
  v16bf b;
#pragma unroll
  for (int e = 0; e < 8; ++e) b[e] = p[e];
#pragma unroll
  for (int e = 0; e < 8; ++e) b[8 + e] = p[16 + e];
  return b;
}

// ---------------- Kernel 1: fused 1x1 conv projections (theta/phi/g) --------
// out(128ch x 4096sp) = W(128x256) @ x(256x4096) + bias, stored bf16
// channel-major (which, reinterpreted row-major 4096x128, IS the reshaped mat).
__global__ __launch_bounds__(256) void proj_kernel(
    const float* __restrict__ x,
    const float* __restrict__ tw, const float* __restrict__ tb,
    const float* __restrict__ pw, const float* __restrict__ pb,
    const float* __restrict__ gw, const float* __restrict__ gb,
    __bf16* __restrict__ thetaB, __bf16* __restrict__ phiB,
    __bf16* __restrict__ gB) {
  const int spBlock = blockIdx.x;  // 0..31
  const int proj = blockIdx.y;     // 0..2
  const int b = blockIdx.z;
  const float* w;
  const float* bias;
  __bf16* dst;
  if (proj == 0)      { w = tw; bias = tb; dst = thetaB; }
  else if (proj == 1) { w = pw; bias = pb; dst = phiB;   }
  else                { w = gw; bias = gb; dst = gB;     }
  dst += (size_t)b * CB * NSP;
  const float* xb = x + (size_t)b * CIN * NSP;
  const int sp0 = spBlock * 128;
  const int tid = threadIdx.x, wave = tid >> 5, lane = tid & 31;

  __shared__ __bf16 ldsXT[128 * 40];  // [sp][k] for one K chunk of 32

  v8f acc[8];
#pragma unroll
  for (int j = 0; j < 8; ++j) acc[j] = (v8f)0.0f;

  for (int k0 = 0; k0 < CIN; k0 += 32) {
    // cooperative load+transpose+convert of x[k0..k0+31][sp0..sp0+127]
    {
      const int cinL = tid >> 3;            // 0..31
      const int so = (tid & 7) * 16;        // 0..112
      const float* src = xb + (size_t)(k0 + cinL) * NSP + sp0 + so;
      if (k0 + 32 < CIN) __builtin_prefetch(src + (size_t)32 * NSP, 0, 1);
#pragma unroll
      for (int e = 0; e < 16; ++e) ldsXT[(so + e) * 40 + cinL] = (__bf16)src[e];
    }
    __syncthreads();
    v16bf a = load_a_f32(w, CIN, wave * 16, k0, lane);
    // software-pipelined B-fragment loads: issue frag j+1 before WMMA j
    v16bf bf = load_b_t(ldsXT, 40, 0, 0, lane);
#pragma unroll
    for (int j = 0; j < 8; ++j) {
      v16bf nxt = bf;
      if (j < 7) nxt = load_b_t(ldsXT, 40, (j + 1) * 16, 0, lane);
      acc[j] = wmma_bf16(a, bf, acc[j]);
      bf = nxt;
    }
    __syncthreads();
  }
  // epilogue: + bias, store channel-major bf16
  const int colL = lane & 15, rg = lane >> 4;
#pragma unroll
  for (int j = 0; j < 8; ++j) {
    const int sp = sp0 + j * 16 + colL;
#pragma unroll
    for (int v = 0; v < 8; ++v) {
      const int ch = wave * 16 + v + 8 * rg;
      dst[(size_t)ch * NSP + sp] = (__bf16)(acc[j][v] + bias[ch]);
    }
  }
}

// ---------------- Kernel 2: per-column softmax stats (online max / sum-exp) --
// scores[n,m] = theta[n,:] . phi[:,m]; softmax axis = n  => column stats.
__global__ __launch_bounds__(256) void colstats_kernel(
    const __bf16* __restrict__ thetaB, const __bf16* __restrict__ phiB,
    float* __restrict__ colmax, float* __restrict__ colrcp) {
  const int mBlock = blockIdx.x;  // 0..31
  const int b = blockIdx.y;
  const __bf16* th = thetaB + (size_t)b * CB * NSP;
  const __bf16* ph = phiB + (size_t)b * CB * NSP;
  const int tid = threadIdx.x, wave = tid >> 5, lane = tid & 31;
  const int m0 = mBlock * 128 + wave * 16;  // this wave's 16 columns
  const int m = m0 + (lane & 15);

  // phi B-fragments (K=Cb=128 -> 4 chunks) held in registers for whole kernel
  v16bf bph[4];
#pragma unroll
  for (int kc = 0; kc < 4; ++kc) {
#pragma unroll
    for (int e = 0; e < 16; ++e) {
      const int k = kc * 32 + (lane >> 4) * 8 + (e < 8 ? e : e + 8);
      bph[kc][e] = ph[(size_t)k * NSP + m];
    }
  }

  __shared__ __bf16 ldsTh[128 * 136];  // theta tile [n'][c], padded stride

  float mx = -3.0e38f, sm = 0.0f;
  for (int n0 = 0; n0 < NSP; n0 += 128) {
    {  // theta rows are contiguous 128-elem chunks at flat n*128
      const int nn = tid >> 1;
      const int half = (tid & 1) * 64;
      const __bf16* src = th + (size_t)(n0 + nn) * CB + half;
      if (n0 + 128 < NSP) __builtin_prefetch(src + (size_t)128 * CB, 0, 1);
      __bf16* d = ldsTh + nn * 136 + half;
#pragma unroll
      for (int i = 0; i < 64; ++i) d[i] = src[i];
    }
    __syncthreads();
    // pipelined A-fragment loads across the (s, kc) sequence
    v16bf a = load_a_bf16(ldsTh, 136, 0, 0, lane);
#pragma unroll
    for (int s = 0; s < 8; ++s) {
      v8f dd = (v8f)0.0f;
#pragma unroll
      for (int kc = 0; kc < 4; ++kc) {
        v16bf an = a;
        if (!(s == 7 && kc == 3)) {
          const int ns = (kc == 3) ? s + 1 : s;
          const int nk = (kc == 3) ? 0 : kc + 1;
          an = load_a_bf16(ldsTh, 136, ns * 16, nk * 32, lane);
        }
        dd = wmma_bf16(a, bph[kc], dd);
        a = an;
      }
      float tmax = dd[0];
#pragma unroll
      for (int v = 1; v < 8; ++v) tmax = fmaxf(tmax, dd[v]);
      tmax = fmaxf(tmax, __shfl_xor(tmax, 16, 32));
      const float nmx = fmaxf(mx, tmax);
      const float scale = __expf(mx - nmx);
      float part = 0.0f;
#pragma unroll
      for (int v = 0; v < 8; ++v) part += __expf(dd[v] - nmx);
      part += __shfl_xor(part, 16, 32);
      sm = sm * scale + part;
      mx = nmx;
    }
    __syncthreads();
  }
  if (lane < 16) {
    colmax[(size_t)b * NSP + m] = mx;
    colrcp[(size_t)b * NSP + m] = 1.0f / sm;
  }
}

// ---------------- Kernel 3: recompute scores, apply softmax, y = P @ g ------
__global__ __launch_bounds__(256) void attn_kernel(
    const __bf16* __restrict__ thetaB, const __bf16* __restrict__ phiB,
    const __bf16* __restrict__ gB, const float* __restrict__ colmax,
    const float* __restrict__ colrcp, __bf16* __restrict__ yB) {
  const int nBlock = blockIdx.x;
  const int b = blockIdx.y;
  const __bf16* th = thetaB + (size_t)b * CB * NSP;
  const __bf16* ph = phiB + (size_t)b * CB * NSP;
  const __bf16* gg = gB + (size_t)b * CB * NSP;
  __bf16* yb = yB + (size_t)b * CB * NSP;
  const float* cmax = colmax + (size_t)b * NSP;
  const float* crcp = colrcp + (size_t)b * NSP;
  const int tid = threadIdx.x, wave = tid >> 5, lane = tid & 31;
  const int n0 = nBlock * 128 + wave * 16;  // wave's 16 output rows

  // theta A-fragments for this wave's rows, kept in registers (K=128)
  v16bf ath[4];
  {
    const __bf16* rp = th + (size_t)(n0 + (lane & 15)) * CB + ((lane >> 4) * 8);
#pragma unroll
    for (int kc = 0; kc < 4; ++kc) {
#pragma unroll
      for (int e = 0; e < 8; ++e) ath[kc][e] = rp[kc * 32 + e];
#pragma unroll
      for (int e = 0; e < 8; ++e) ath[kc][8 + e] = rp[kc * 32 + 16 + e];
    }
  }

  v8f yacc[8];
#pragma unroll
  for (int j = 0; j < 8; ++j) yacc[j] = (v8f)0.0f;

  __shared__ __bf16 ldsPh[64 * 136];   // phi tile transposed: [m'][c]
  __shared__ __bf16 ldsG[128 * 72];    // g tile transposed: [c][m']
  __shared__ __bf16 ldsP[8][16 * 40];  // per-wave 16x32 P tile (D->A transpose)

  for (int m0 = 0; m0 < NSP; m0 += 64) {
    {  // stage phi[c][m0..m0+63] -> ldsPh[m'][c]
      const int c = tid >> 1;
      const int half = (tid & 1) * 32;
      const __bf16* src = ph + (size_t)c * NSP + m0 + half;
      if (m0 + 64 < NSP) __builtin_prefetch(src + 64, 0, 1);
#pragma unroll
      for (int i = 0; i < 32; ++i) ldsPh[(half + i) * 136 + c] = src[i];
    }
    {  // stage g rows m0..m0+63 (flat m*128, contiguous) -> ldsG[c][m']
      const int mm = tid >> 2;
      const int q = (tid & 3) * 32;
      const __bf16* src = gg + (size_t)(m0 + mm) * CB + q;
      if (m0 + 64 < NSP) __builtin_prefetch(src + (size_t)64 * CB, 0, 1);
#pragma unroll
      for (int i = 0; i < 32; ++i) ldsG[(q + i) * 72 + mm] = src[i];
    }
    __syncthreads();

#pragma unroll
    for (int pair = 0; pair < 2; ++pair) {  // 32 m-values per pair
#pragma unroll
      for (int jj = 0; jj < 2; ++jj) {
        const int mloc = pair * 32 + jj * 16;
        v8f dd = (v8f)0.0f;
        v16bf bf = load_b_t(ldsPh, 136, mloc, 0, lane);
#pragma unroll
        for (int kc = 0; kc < 4; ++kc) {
          v16bf nxt = bf;
          if (kc < 3) nxt = load_b_t(ldsPh, 136, mloc, (kc + 1) * 32, lane);
          dd = wmma_bf16(ath[kc], bf, dd);
          bf = nxt;
        }
        const int m = m0 + mloc + (lane & 15);
        const float cm = cmax[m];
        const float cr = crcp[m];
        // write P tile bf16 to per-wave LDS (D layout -> row-major 16x32)
#pragma unroll
        for (int v = 0; v < 8; ++v) {
          const int r = v + 8 * (lane >> 4);
          ldsP[wave][r * 40 + jj * 16 + (lane & 15)] =
              (__bf16)(__expf(dd[v] - cm) * cr);
        }
      }
      // P as A-matrix fragment (16 rows n x 32 k=m); same-wave DS ordering
      v16bf pa = load_a_bf16(ldsP[wave], 40, 0, 0, lane);
      v16bf bg = load_b_t(ldsG, 72, 0, pair * 32, lane);
#pragma unroll
      for (int cc = 0; cc < 8; ++cc) {
        v16bf nxt = bg;
        if (cc < 7) nxt = load_b_t(ldsG, 72, (cc + 1) * 16, pair * 32, lane);
        yacc[cc] = wmma_bf16(pa, bg, yacc[cc]);
        bg = nxt;
      }
    }
    __syncthreads();
  }

  // store y (matrix view flat n*128+c == channel-major buffer)
#pragma unroll
  for (int cc = 0; cc < 8; ++cc) {
    const int c = cc * 16 + (lane & 15);
#pragma unroll
    for (int v = 0; v < 8; ++v) {
      const int n = n0 + v + 8 * (lane >> 4);
      yb[(size_t)n * CB + c] = (__bf16)yacc[cc][v];
    }
  }
}

// ---------------- Kernel 4: final 1x1 conv (256x128) + bias + residual ------
__global__ __launch_bounds__(256) void outconv_kernel(
    const __bf16* __restrict__ yB, const float* __restrict__ Ww,
    const float* __restrict__ Wb, const float* __restrict__ x,
    float* __restrict__ out) {
  const int spBlock = blockIdx.x;  // 0..31
  const int coB = blockIdx.y;      // 0..1
  const int b = blockIdx.z;
  const __bf16* y = yB + (size_t)b * CB * NSP;
  const int sp0 = spBlock * 128;
  const int tid = threadIdx.x, wave = tid >> 5, lane = tid & 31;
  const int co0 = coB * 128 + wave * 16;

  __shared__ __bf16 ldsYT[128 * 136];  // [sp][o]
  {
    const int o = tid >> 1;
    const int half = (tid & 1) * 64;
    const __bf16* src = y + (size_t)o * NSP + sp0 + half;
#pragma unroll
    for (int i = 0; i < 64; ++i) ldsYT[(half + i) * 136 + o] = src[i];
  }
  __syncthreads();

  v8f acc[8];
#pragma unroll
  for (int j = 0; j < 8; ++j) acc[j] = (v8f)0.0f;
  v16bf a = load_a_f32(Ww, CB, co0, 0, lane);
#pragma unroll
  for (int kc = 0; kc < 4; ++kc) {
    v16bf an = a;
    if (kc < 3) an = load_a_f32(Ww, CB, co0, (kc + 1) * 32, lane);
    v16bf bf = load_b_t(ldsYT, 136, 0, kc * 32, lane);
#pragma unroll
    for (int j = 0; j < 8; ++j) {
      v16bf nxt = bf;
      if (j < 7) nxt = load_b_t(ldsYT, 136, (j + 1) * 16, kc * 32, lane);
      acc[j] = wmma_bf16(a, bf, acc[j]);
      bf = nxt;
    }
    a = an;
  }
#pragma unroll
  for (int j = 0; j < 8; ++j) {
    const int sp = sp0 + j * 16 + (lane & 15);
#pragma unroll
    for (int v = 0; v < 8; ++v) {
      const int co = co0 + v + 8 * (lane >> 4);
      const size_t idx = ((size_t)b * CIN + co) * NSP + sp;
      out[idx] = acc[j][v] + Wb[co] + x[idx];
    }
  }
}

extern "C" void kernel_launch(void* const* d_in, const int* in_sizes, int n_in,
                              void* d_out, int out_size, void* d_ws,
                              size_t ws_size, hipStream_t stream) {
  const float* x  = (const float*)d_in[0];
  const float* tw = (const float*)d_in[1];
  const float* tb = (const float*)d_in[2];
  const float* pw = (const float*)d_in[3];
  const float* pb = (const float*)d_in[4];
  const float* gw = (const float*)d_in[5];
  const float* gb = (const float*)d_in[6];
  const float* Ww = (const float*)d_in[7];
  const float* Wb = (const float*)d_in[8];
  float* out = (float*)d_out;

  char* ws = (char*)d_ws;
  const size_t projBytes = (size_t)BATCH * CB * NSP * 2;  // bf16
  __bf16* thetaB = (__bf16*)(ws);
  __bf16* phiB   = (__bf16*)(ws + projBytes);
  __bf16* gB     = (__bf16*)(ws + 2 * projBytes);
  __bf16* yB     = (__bf16*)(ws + 3 * projBytes);
  float* colmax  = (float*)(ws + 4 * projBytes);
  float* colrcp  = (float*)(ws + 4 * projBytes + (size_t)BATCH * NSP * 4);

  dim3 blk(256);
  proj_kernel<<<dim3(NSP / 128, 3, BATCH), blk, 0, stream>>>(
      x, tw, tb, pw, pb, gw, gb, thetaB, phiB, gB);
  colstats_kernel<<<dim3(NSP / 128, BATCH), blk, 0, stream>>>(
      thetaB, phiB, colmax, colrcp);
  attn_kernel<<<dim3(NSP / 128, BATCH), blk, 0, stream>>>(
      thetaB, phiB, gB, colmax, colrcp, yB);
  outconv_kernel<<<dim3(NSP / 128, 2, BATCH), blk, 0, stream>>>(
      yB, Ww, Wb, x, out);
}
